// ArcMarginLoss_68229850464345
// MI455X (gfx1250) — compile-verified, matrix operand order checked by reading, not run on hardware
//
#include <hip/hip_runtime.h>

// ---------------------------------------------------------------------------
// ArcFace margin loss, fused.  MI455X (gfx1250, wave32, WMMA, async-to-LDS).
// ---------------------------------------------------------------------------

typedef __attribute__((ext_vector_type(16))) __bf16 v16bf;
typedef __attribute__((ext_vector_type(8)))  __bf16 v8bf;
typedef __attribute__((ext_vector_type(4)))  __bf16 v4bf;
typedef __attribute__((ext_vector_type(8)))  float  v8f;

#define N_ROWS 8192
#define D_DIM  512
#define C_CLS  32000
#define S_SC   16.0f
#define COS_M  0.9800665778412416f   // cos(0.2)
#define SIN_M  0.19866933079506122f  // sin(0.2)
#define EPS_C  1e-7f

#define BM 64          // rows per block
#define BN 128         // cols per c-tile
#define BK 32          // k per WMMA step
#define LDK 40         // padded LDS stride (80B = 20 banks -> conflict-free b128)
#define CSPLIT 5       // C split across gridDim.y
#define CCHUNK (C_CLS / CSPLIT)   // 6400 = 50 tiles of 128

// workspace layout (bytes)
#define OFF_XHI   ((size_t)0)
#define OFF_XLO   ((size_t)8388608)        // 8192*512*2
#define OFF_WHI   ((size_t)16777216)
#define OFF_WLO   ((size_t)49545216)       // + 32000*512*2
#define OFF_COST  ((size_t)82313216)
#define OFF_PST   ((size_t)82345984)       // 8192*5*2 floats = 327680
#define OFF_NLL   ((size_t)82673664)

// ---------------------------------------------------------------------------
// CDNA5 async copy, GVS mode: SGPR-pair base + 32-bit VGPR byte offset.
// 16B global -> LDS per lane, tracked by ASYNCcnt.  lds_off is the LDS byte
// address (generic LDS pointer truncated to its low 32 bits).
// ---------------------------------------------------------------------------
__device__ __forceinline__ void async_gvs(unsigned lds_off, unsigned goff,
                                          const void* sbase) {
    asm volatile("global_load_async_to_lds_b128 %0, %1, %2"
                 :: "v"(lds_off), "v"(goff), "s"(sbase) : "memory");
}
__device__ __forceinline__ void wait_async0() {
#if __has_builtin(__builtin_amdgcn_s_wait_asynccnt)
    __builtin_amdgcn_s_wait_asynccnt(0);
#else
    asm volatile("s_wait_asynccnt 0" ::: "memory");
#endif
}

// ---------------------------------------------------------------------------
// Kernel 1: L2-normalize rows of a [rows,512] fp32 matrix and split each
// normalized value into bf16 hi + bf16 lo (x ~= hi + lo).  One wave per row.
// ---------------------------------------------------------------------------
__global__ __launch_bounds__(256)
void prep_rows(const float* __restrict__ src, __bf16* __restrict__ hi,
               __bf16* __restrict__ lo, int nrows) {
    const int wid  = (blockIdx.x * 256 + threadIdx.x) >> 5;
    const int lane = threadIdx.x & 31;
    if (wid >= nrows) return;

    const float4* p = (const float4*)(src + (size_t)wid * D_DIM);
    float4 v[4];
    float ss = 0.f;
#pragma unroll
    for (int j = 0; j < 4; ++j) {
        v[j] = p[j * 32 + lane];
        ss += v[j].x * v[j].x + v[j].y * v[j].y + v[j].z * v[j].z + v[j].w * v[j].w;
    }
#pragma unroll
    for (int off = 16; off > 0; off >>= 1) ss += __shfl_xor(ss, off);
    const float rn = 1.f / fmaxf(sqrtf(ss), 1e-12f);

#pragma unroll
    for (int j = 0; j < 4; ++j) {
        float f[4] = {v[j].x * rn, v[j].y * rn, v[j].z * rn, v[j].w * rn};
        v4bf h, l;
#pragma unroll
        for (int k = 0; k < 4; ++k) {
            __bf16 hb = (__bf16)f[k];
            h[k] = hb;
            l[k] = (__bf16)(f[k] - (float)hb);
        }
        const size_t idx = (size_t)wid * D_DIM + (j * 32 + lane) * 4;
        *(v4bf*)(hi + idx) = h;
        *(v4bf*)(lo + idx) = l;
    }
}

// ---------------------------------------------------------------------------
// Kernel 2: exact fp32 target cosine: cos_t[n] = <x_n, w_label[n]> / (|x||w|).
// ---------------------------------------------------------------------------
__global__ __launch_bounds__(256)
void cos_target(const float* __restrict__ x, const float* __restrict__ w,
                const int* __restrict__ labels, float* __restrict__ cos_t) {
    const int wid  = (blockIdx.x * 256 + threadIdx.x) >> 5;
    const int lane = threadIdx.x & 31;
    if (wid >= N_ROWS) return;
    const int lab = labels[wid];

    const float4* px = (const float4*)(x + (size_t)wid * D_DIM);
    const float4* pw = (const float4*)(w + (size_t)lab * D_DIM);
    float sxw = 0.f, sxx = 0.f, sww = 0.f;
#pragma unroll
    for (int j = 0; j < 4; ++j) {
        float4 a = px[j * 32 + lane];
        float4 b = pw[j * 32 + lane];
        sxw += a.x * b.x + a.y * b.y + a.z * b.z + a.w * b.w;
        sxx += a.x * a.x + a.y * a.y + a.z * a.z + a.w * a.w;
        sww += b.x * b.x + b.y * b.y + b.z * b.z + b.w * b.w;
    }
#pragma unroll
    for (int off = 16; off > 0; off >>= 1) {
        sxw += __shfl_xor(sxw, off);
        sxx += __shfl_xor(sxx, off);
        sww += __shfl_xor(sww, off);
    }
    if (lane == 0)
        cos_t[wid] = sxw / (fmaxf(sqrtf(sxx), 1e-12f) * fmaxf(sqrtf(sww), 1e-12f));
}

// ---------------------------------------------------------------------------
// LDS fragment loader: 16-bit A/B WMMA fragment per CDNA5 ISA layout.
//   lanes 0-15 : K = k0+0..7  and k0+16..23
//   lanes 16-31: K = k0+8..15 and k0+24..31
// ---------------------------------------------------------------------------
__device__ __forceinline__ v16bf frag16(const __bf16* rowptr, int half) {
    v8bf c0 = *(const v8bf*)(rowptr + half * 8);
    v8bf c1 = *(const v8bf*)(rowptr + 16 + half * 8);
    v16bf r;
#pragma unroll
    for (int i = 0; i < 8; ++i) { r[i] = c0[i]; r[i + 8] = c1[i]; }
    return r;
}

#define WMMA_BF16(A, B, Cacc) \
    __builtin_amdgcn_wmma_f32_16x16x32_bf16(false, (A), false, (B), (short)0, (Cacc), false, false)

// ---------------------------------------------------------------------------
// Kernel 3: split-bf16 WMMA GEMM (cosine) fused with online softmax over a
// CCHUNK-wide class slice.  Block = 64 rows x 128 cols per tile, 8 waves as
// 2(mi) x 4(ni); wave tile = 32 rows x 32 cols = 4 accumulators, 12 WMMA and
// 16 ds_load_b128 per k-step.  Staging via global_load_async_to_lds_b128
// (GVS: SGPR base + u32 offset) double-buffered, one barrier per k-step.
// ---------------------------------------------------------------------------
__global__ __launch_bounds__(256)
void gemm_softmax(const __bf16* __restrict__ xhi, const __bf16* __restrict__ xlo,
                  const __bf16* __restrict__ whi, const __bf16* __restrict__ wlo,
                  float* __restrict__ pstats) {
    __shared__ __bf16 sAhi[2][BM * LDK];
    __shared__ __bf16 sAlo[2][BM * LDK];
    __shared__ __bf16 sBhi[2][BN * LDK];
    __shared__ __bf16 sBlo[2][BN * LDK];
    __shared__ float  sStats[BM][4][2];

    const int tid  = threadIdx.x;
    const int wave = tid >> 5;
    const int lane = tid & 31;
    const int half = lane >> 4;
    const int l    = lane & 15;
    const int mi   = wave & 1;   // row half: 32 rows
    const int ni   = wave >> 1;  // col quarter: 32 cols

    const int rowbase = blockIdx.x * BM;
    const int cbase   = blockIdx.y * CCHUNK;

    // staging map: thread t moves 16B chunks at (row = t>>2, ko = (t&3)*8)
    // for A(hi,lo) and B rows (row, row+64) (hi,lo): 6 async b128 per k-step.
    const int srow = tid >> 2;          // 0..63
    const int sko  = (tid & 3) * 8;     // 0,8,16,24

    // u32 global byte offsets (hi/lo arrays share the same layout)
    const unsigned offA = (unsigned)(rowbase + srow) * (D_DIM * 2) + sko * 2;
    unsigned offB0 = (unsigned)(cbase + srow) * (D_DIM * 2) + sko * 2;
    unsigned offB1 = offB0 + 64u * (D_DIM * 2);

    // LDS byte offsets for both buffers
    const unsigned oAhi[2] = {(unsigned)(uintptr_t)&sAhi[0][srow * LDK + sko],
                              (unsigned)(uintptr_t)&sAhi[1][srow * LDK + sko]};
    const unsigned oAlo[2] = {(unsigned)(uintptr_t)&sAlo[0][srow * LDK + sko],
                              (unsigned)(uintptr_t)&sAlo[1][srow * LDK + sko]};
    const unsigned oBhi0[2] = {(unsigned)(uintptr_t)&sBhi[0][srow * LDK + sko],
                               (unsigned)(uintptr_t)&sBhi[1][srow * LDK + sko]};
    const unsigned oBhi1[2] = {(unsigned)(uintptr_t)&sBhi[0][(srow + 64) * LDK + sko],
                               (unsigned)(uintptr_t)&sBhi[1][(srow + 64) * LDK + sko]};
    const unsigned oBlo0[2] = {(unsigned)(uintptr_t)&sBlo[0][srow * LDK + sko],
                               (unsigned)(uintptr_t)&sBlo[1][srow * LDK + sko]};
    const unsigned oBlo1[2] = {(unsigned)(uintptr_t)&sBlo[0][(srow + 64) * LDK + sko],
                               (unsigned)(uintptr_t)&sBlo[1][(srow + 64) * LDK + sko]};

    float rmax[16], rsum[16];
#pragma unroll
    for (int i = 0; i < 16; ++i) { rmax[i] = -3.0e38f; rsum[i] = 0.f; }

#pragma unroll 1
    for (int ct = 0; ct < CCHUNK; ct += BN) {
        v8f a00 = {}, a01 = {}, a10 = {}, a11 = {};  // [mt][nt]

        auto stage = [&](int buf, unsigned kbyte) {
            const unsigned vA  = offA + kbyte;    // 3 u32 adds total
            const unsigned vB0 = offB0 + kbyte;
            const unsigned vB1 = offB1 + kbyte;
            async_gvs(oAhi[buf], vA, xhi);
            async_gvs(oAlo[buf], vA, xlo);
            async_gvs(oBhi0[buf], vB0, whi);
            async_gvs(oBhi1[buf], vB1, whi);
            async_gvs(oBlo0[buf], vB0, wlo);
            async_gvs(oBlo1[buf], vB1, wlo);
        };

        auto compute = [&](int cur) {
            const __bf16* A_hi = &sAhi[cur][0];
            const __bf16* A_lo = &sAlo[cur][0];
            const __bf16* B_hi = &sBhi[cur][0];
            const __bf16* B_lo = &sBlo[cur][0];
            v16bf ah0 = frag16(A_hi + (mi * 32 + l) * LDK, half);
            v16bf al0 = frag16(A_lo + (mi * 32 + l) * LDK, half);
            v16bf ah1 = frag16(A_hi + (mi * 32 + 16 + l) * LDK, half);
            v16bf al1 = frag16(A_lo + (mi * 32 + 16 + l) * LDK, half);
            {   // B subtile 0
                v16bf bh = frag16(B_hi + (ni * 32 + l) * LDK, half);
                v16bf bl = frag16(B_lo + (ni * 32 + l) * LDK, half);
                a00 = WMMA_BF16(ah0, bh, a00);
                a00 = WMMA_BF16(ah0, bl, a00);
                a00 = WMMA_BF16(al0, bh, a00);
                a10 = WMMA_BF16(ah1, bh, a10);
                a10 = WMMA_BF16(ah1, bl, a10);
                a10 = WMMA_BF16(al1, bh, a10);
            }
            {   // B subtile 1
                v16bf bh = frag16(B_hi + (ni * 32 + 16 + l) * LDK, half);
                v16bf bl = frag16(B_lo + (ni * 32 + 16 + l) * LDK, half);
                a01 = WMMA_BF16(ah0, bh, a01);
                a01 = WMMA_BF16(ah0, bl, a01);
                a01 = WMMA_BF16(al0, bh, a01);
                a11 = WMMA_BF16(ah1, bh, a11);
                a11 = WMMA_BF16(ah1, bl, a11);
                a11 = WMMA_BF16(al1, bh, a11);
            }
        };

        // prologue: k-chunk 0 into buffer 0
        stage(0, 0u);
        wait_async0();
        __syncthreads();

#pragma unroll 2
        for (int kt = 0; kt < D_DIM / BK - 1; ++kt) {
            const int cur = kt & 1;
            stage(cur ^ 1, (unsigned)(kt + 1) * (BK * 2));  // freed by barrier @ kt-1
            compute(cur);
            wait_async0();
            __syncthreads();
        }
        compute(1);                           // kt = 15, buffer 1, no prefetch

        // advance B column offsets to next c-tile
        offB0 += BN * (D_DIM * 2);
        offB1 += BN * (D_DIM * 2);

        // online softmax update: acc(mt,nt)[i] is row mi*32+mt*16+half*8+i
#pragma unroll
        for (int mt = 0; mt < 2; ++mt) {
#pragma unroll
            for (int i = 0; i < 8; ++i) {
                const int si = mt * 8 + i;
                const float v0 = (mt ? a10[i] : a00[i]) * S_SC;
                const float v1 = (mt ? a11[i] : a01[i]) * S_SC;
                const float nm = fmaxf(rmax[si], fmaxf(v0, v1));
                rsum[si] = rsum[si] * __expf(rmax[si] - nm) + __expf(v0 - nm) + __expf(v1 - nm);
                rmax[si] = nm;
            }
        }
    }

    // reduce running stats across the 16 lanes of each half-wave
#pragma unroll
    for (int si = 0; si < 16; ++si) {
        float m = rmax[si], s = rsum[si];
#pragma unroll
        for (int off = 1; off < 16; off <<= 1) {
            const float om = __shfl_xor(m, off);
            const float os = __shfl_xor(s, off);
            const float nm = fmaxf(m, om);
            s = s * __expf(m - nm) + os * __expf(om - nm);
            m = nm;
        }
        if (l == 0) {
            const int r = mi * 32 + (si >> 3) * 16 + half * 8 + (si & 7);
            sStats[r][ni][0] = m;
            sStats[r][ni][1] = s;
        }
    }
    __syncthreads();

    // merge the 4 column-waves covering the same rows; write partial stats
    if (tid < BM) {
        float m = sStats[tid][0][0], s = sStats[tid][0][1];
#pragma unroll
        for (int k = 1; k < 4; ++k) {
            const float om = sStats[tid][k][0], os = sStats[tid][k][1];
            const float nm = fmaxf(m, om);
            s = s * __expf(m - nm) + os * __expf(om - nm);
            m = nm;
        }
        float* p = pstats + (size_t)(rowbase + tid) * (CSPLIT * 2) + blockIdx.y * 2;
        p[0] = m;
        p[1] = s;
    }
}

// ---------------------------------------------------------------------------
// Kernel 4: merge CSPLIT partial LSE stats, apply ArcFace margin swap for the
// target class, emit per-row NLL.
// ---------------------------------------------------------------------------
__global__ __launch_bounds__(256)
void finalize_nll(const float* __restrict__ pstats, const float* __restrict__ cos_t,
                  float* __restrict__ nll) {
    const int n = blockIdx.x * 256 + threadIdx.x;
    if (n >= N_ROWS) return;
    const float* p = pstats + (size_t)n * (CSPLIT * 2);
    float m = p[0], s = p[1];
#pragma unroll
    for (int k = 1; k < CSPLIT; ++k) {
        const float om = p[k * 2], os = p[k * 2 + 1];
        const float nm = fmaxf(m, om);
        s = s * __expf(m - nm) + os * __expf(om - nm);
        m = nm;
    }
    const float ct = cos_t[n];
    const float c  = fminf(fmaxf(ct, -1.f + EPS_C), 1.f - EPS_C);
    // cos(theta + M) = c*cosM - sqrt(1-c^2)*sinM   (exact, no trig calls)
    const float marg = S_SC * (c * COS_M - sqrtf(1.f - c * c) * SIN_M);
    const float lt   = S_SC * ct;
    const float nm   = fmaxf(m, marg);
    float s2 = s * __expf(m - nm) - __expf(lt - nm) + __expf(marg - nm);
    s2 = fmaxf(s2, 1e-37f);
    nll[n] = nm + __logf(s2) - marg;
}

// ---------------------------------------------------------------------------
// Kernel 5: deterministic mean over N_ROWS NLL values -> scalar output.
// ---------------------------------------------------------------------------
__global__ __launch_bounds__(256)
void reduce_mean(const float* __restrict__ nll, float* __restrict__ out) {
    __shared__ float sbuf[256];
    float s = 0.f;
    for (int i = threadIdx.x; i < N_ROWS; i += 256) s += nll[i];
    sbuf[threadIdx.x] = s;
    __syncthreads();
    for (int off = 128; off > 0; off >>= 1) {
        if (threadIdx.x < off) sbuf[threadIdx.x] += sbuf[threadIdx.x + off];
        __syncthreads();
    }
    if (threadIdx.x == 0) out[0] = sbuf[0] * (1.0f / N_ROWS);
}

// ---------------------------------------------------------------------------
extern "C" void kernel_launch(void* const* d_in, const int* in_sizes, int n_in,
                              void* d_out, int out_size, void* d_ws, size_t ws_size,
                              hipStream_t stream) {
    const float* prev   = (const float*)d_in[0];   // [8192, 512]
    const float* weight = (const float*)d_in[1];   // [32000, 512]
    const int*   labels = (const int*)d_in[2];     // [8192]
    float* out = (float*)d_out;

    char* ws = (char*)d_ws;
    __bf16* xhi = (__bf16*)(ws + OFF_XHI);
    __bf16* xlo = (__bf16*)(ws + OFF_XLO);
    __bf16* whi = (__bf16*)(ws + OFF_WHI);
    __bf16* wlo = (__bf16*)(ws + OFF_WLO);
    float* cost   = (float*)(ws + OFF_COST);
    float* pstats = (float*)(ws + OFF_PST);
    float* nll    = (float*)(ws + OFF_NLL);

    prep_rows<<<N_ROWS / 8, 256, 0, stream>>>(prev, xhi, xlo, N_ROWS);
    prep_rows<<<C_CLS / 8, 256, 0, stream>>>(weight, whi, wlo, C_CLS);
    cos_target<<<N_ROWS / 8, 256, 0, stream>>>(prev, weight, labels, cost);

    dim3 grid(N_ROWS / BM, CSPLIT, 1);
    gemm_softmax<<<grid, 256, 0, stream>>>(xhi, xlo, whi, wlo, pstats);

    finalize_nll<<<N_ROWS / 256, 256, 0, stream>>>(pstats, cost, nll);
    reduce_mean<<<1, 256, 0, stream>>>(nll, out);
}
